// ScaledDotProduct_21723944583211
// MI455X (gfx1250) — matrix operand order repaired and checked by reading, compiled, not run on hardware
//
#include <hip/hip_runtime.h>
#include <math.h>

typedef __attribute__((ext_vector_type(16))) _Float16 v16h;
typedef __attribute__((ext_vector_type(2)))  _Float16 v2h;
typedef __attribute__((ext_vector_type(8)))  float    v8f;

#define SEQ      2048
#define DH       128
#define MT       128         // Q-tile rows per block (4 waves x 32 rows)
#define NT       64          // K/V-tile rows per iteration
#define NTHREADS 128         // 4 waves (wave32)
#define QSTR     136         // 128 + 8 pad (halfs) for Q staging
#define KSTR     136
#define VSTR     72          // 64 + 8 pad (halfs), V stored transposed [d][k]
#define PSTR     68          // 64 + 4 pad (halfs); 4 waves*32 rows*68 = 8704 = stage size

__global__ __launch_bounds__(NTHREADS)
void flash_attn_wmma_f16(const float* __restrict__ Q,
                         const float* __restrict__ K,
                         const float* __restrict__ V,
                         float* __restrict__ O)
{
    __shared__ _Float16 sStage[64 * QSTR];  // Q staging (2 passes); then P scratch
    __shared__ _Float16 sK [NT * KSTR];     // K tile, row-major
    __shared__ _Float16 sVt[DH * VSTR];     // V tile transposed: [d][k]

    const int tid  = threadIdx.x;
    const int wave = tid >> 5;
    const int lane = tid & 31;
    const int grp  = lane >> 4;      // 0: lanes 0-15, 1: lanes 16-31
    const int ln   = lane & 15;
    const int bh   = blockIdx.y;
    const int q0   = blockIdx.x * MT;
    const int wr   = wave * 32;      // this wave's 32-row slice of the Q tile

    const float scale = 0.08838834764831845f;  // 1/sqrt(128)

    const size_t bhoff = (size_t)bh * SEQ * DH;
    const float* Qb = Q + bhoff;
    const float* Kb = K + bhoff;
    const float* Vb = V + bhoff;
    float*       Ob = O + bhoff;

    // ---- stage Q tile (two 64-row passes through sStage), gather A fragments ----
    // ISA 16-bit A 16x32 layout: lanes 0-15 hold K 0-7/16-23, lanes 16-31 K 8-15/24-31.
    v16h aq[2][4];
    for (int chunk = 0; chunk < 2; ++chunk) {
        for (int i = tid; i < 64 * (DH / 4); i += NTHREADS) {
            const int row = i >> 5, c4 = i & 31;
            const float4 v = *(const float4*)(Qb + (size_t)(q0 + chunk * 64 + row) * DH + c4 * 4);
            _Float16* dst = &sStage[row * QSTR + c4 * 4];
            dst[0] = (_Float16)(v.x * scale);
            dst[1] = (_Float16)(v.y * scale);
            dst[2] = (_Float16)(v.z * scale);
            dst[3] = (_Float16)(v.w * scale);
        }
        __syncthreads();
        if ((wr >> 6) == chunk) {
            const int lr = wr - chunk * 64;
            #pragma unroll
            for (int mb = 0; mb < 2; ++mb)
                #pragma unroll
                for (int c = 0; c < 4; ++c)
                    #pragma unroll
                    for (int p = 0; p < 8; ++p) {
                        const int k = 32 * c + 2 * p + 8 * grp + ((p >= 4) ? 8 : 0);
                        const v2h h = *(const v2h*)&sStage[(lr + mb * 16 + ln) * QSTR + k];
                        aq[mb][c][2 * p]     = h.x;
                        aq[mb][c][2 * p + 1] = h.y;
                    }
        }
        __syncthreads();
    }

    v8f   oacc[2][8] = {};          // 32 rows x 128 cols of O (f32 accum)
    float m_i[2][8], l_i[2][8];
    #pragma unroll
    for (int mb = 0; mb < 2; ++mb)
        #pragma unroll
        for (int r = 0; r < 8; ++r) { m_i[mb][r] = -1e30f; l_i[mb][r] = 0.0f; }

    _Float16* sP = sStage + wave * 32 * PSTR;   // per-wave P scratch (32 x 64)

    for (int t = 0; t < SEQ / NT; ++t) {
        const int k0 = t * NT;

        // ---- cooperative K tile load (f32 -> f16) ----
        for (int i = tid; i < NT * (DH / 4); i += NTHREADS) {
            const int row = i >> 5, c4 = i & 31;
            const float4 v = *(const float4*)(Kb + (size_t)(k0 + row) * DH + c4 * 4);
            _Float16* dst = &sK[row * KSTR + c4 * 4];
            dst[0] = (_Float16)v.x; dst[1] = (_Float16)v.y;
            dst[2] = (_Float16)v.z; dst[3] = (_Float16)v.w;
        }
        // ---- cooperative V tile load, transposed into sVt[d][k] ----
        for (int i = tid; i < NT * (DH / 4); i += NTHREADS) {
            const int row = i >> 5, c4 = i & 31;
            const float4 v = *(const float4*)(Vb + (size_t)(k0 + row) * DH + c4 * 4);
            sVt[(c4 * 4 + 0) * VSTR + row] = (_Float16)v.x;
            sVt[(c4 * 4 + 1) * VSTR + row] = (_Float16)v.y;
            sVt[(c4 * 4 + 2) * VSTR + row] = (_Float16)v.z;
            sVt[(c4 * 4 + 3) * VSTR + row] = (_Float16)v.w;
        }
        __syncthreads();

        // ---- S = (Q*scale) @ K^T : each B fragment feeds TWO WMMAs (mb=0,1) ----
        v8f sacc[2][4];
        #pragma unroll
        for (int nb = 0; nb < 4; ++nb) {
            v8f a0 = {}, a1 = {};
            #pragma unroll
            for (int c = 0; c < 4; ++c) {
                v16h b;   // B[k][n] = K[nb*16+n][32c+k]; table k order (matches A)
                #pragma unroll
                for (int p = 0; p < 8; ++p) {
                    const v2h h = *(const v2h*)&sK[(nb * 16 + ln) * KSTR
                                                   + 32 * c + 8 * grp + 2 * p + ((p >= 4) ? 8 : 0)];
                    b[2 * p] = h.x; b[2 * p + 1] = h.y;
                }
                a0 = __builtin_amdgcn_wmma_f32_16x16x32_f16(
                         false, aq[0][c], false, b, (short)0, a0, false, false);
                a1 = __builtin_amdgcn_wmma_f32_16x16x32_f16(
                         false, aq[1][c], false, b, (short)0, a1, false, false);
            }
            sacc[0][nb] = a0;
            sacc[1][nb] = a1;
        }

        // ---- online softmax (C layout: row = r + 8*grp, col = nb*16 + ln) ----
        #pragma unroll
        for (int mb = 0; mb < 2; ++mb) {
            float mnew[8], corr[8];
            #pragma unroll
            for (int r = 0; r < 8; ++r) {
                float v = fmaxf(fmaxf(sacc[mb][0][r], sacc[mb][1][r]),
                                fmaxf(sacc[mb][2][r], sacc[mb][3][r]));
                #pragma unroll
                for (int off = 1; off < 16; off <<= 1)
                    v = fmaxf(v, __shfl_xor(v, off, 32));
                const float mn = fmaxf(m_i[mb][r], v);
                mnew[r] = mn;
                corr[r] = __expf(m_i[mb][r] - mn);
            }
            float rsum[8];
            #pragma unroll
            for (int r = 0; r < 8; ++r) rsum[r] = 0.0f;
            #pragma unroll
            for (int nb = 0; nb < 4; ++nb)
                #pragma unroll
                for (int r = 0; r < 8; ++r) {
                    const float pv = __expf(sacc[mb][nb][r] - mnew[r]);
                    rsum[r] += pv;
                    sP[(mb * 16 + r + 8 * grp) * PSTR + nb * 16 + ln] = (_Float16)pv;
                }
            #pragma unroll
            for (int r = 0; r < 8; ++r) {
                float v = rsum[r];
                #pragma unroll
                for (int off = 1; off < 16; off <<= 1)
                    v += __shfl_xor(v, off, 32);
                l_i[mb][r] = l_i[mb][r] * corr[r] + v;
                m_i[mb][r] = mnew[r];
            }
            #pragma unroll
            for (int d = 0; d < 8; ++d)
                #pragma unroll
                for (int r = 0; r < 8; ++r)
                    oacc[mb][d][r] *= corr[r];
        }

        // ---- O += P @ V : each V B-fragment feeds TWO WMMAs (mb=0,1) ----
        #pragma unroll
        for (int kc = 0; kc < 2; ++kc) {
            v16h ap[2];   // A fragments of P (same-wave LDS, DS ops are in-order)
            #pragma unroll
            for (int mb = 0; mb < 2; ++mb)
                #pragma unroll
                for (int p = 0; p < 8; ++p) {
                    const int k = 32 * kc + 2 * p + 8 * grp + ((p >= 4) ? 8 : 0);
                    const v2h h = *(const v2h*)&sP[(mb * 16 + ln) * PSTR + k];
                    ap[mb][2 * p] = h.x; ap[mb][2 * p + 1] = h.y;
                }
            #pragma unroll
            for (int d = 0; d < 8; ++d) {
                v16h b;   // B[k][n] = V[k0+32kc+k][d*16+n] = sVt[d*16+n][32kc+k]
                #pragma unroll
                for (int p = 0; p < 8; ++p) {
                    const v2h h = *(const v2h*)&sVt[(d * 16 + ln) * VSTR
                                                    + 32 * kc + 8 * grp + 2 * p + ((p >= 4) ? 8 : 0)];
                    b[2 * p] = h.x; b[2 * p + 1] = h.y;
                }
                oacc[0][d] = __builtin_amdgcn_wmma_f32_16x16x32_f16(
                                 false, ap[0], false, b, (short)0, oacc[0][d], false, false);
                oacc[1][d] = __builtin_amdgcn_wmma_f32_16x16x32_f16(
                                 false, ap[1], false, b, (short)0, oacc[1][d], false, false);
            }
        }
        __syncthreads();   // done with sK/sVt before next tile overwrites them
    }

    // ---- epilogue: O /= l, write out (C layout) ----
    #pragma unroll
    for (int mb = 0; mb < 2; ++mb)
        #pragma unroll
        for (int r = 0; r < 8; ++r) {
            const float inv = 1.0f / l_i[mb][r];
            const size_t row = (size_t)(q0 + wr + mb * 16 + r + 8 * grp) * DH;
            #pragma unroll
            for (int d = 0; d < 8; ++d)
                Ob[row + d * 16 + ln] = oacc[mb][d][r] * inv;
        }
}

extern "C" void kernel_launch(void* const* d_in, const int* in_sizes, int n_in,
                              void* d_out, int out_size, void* d_ws, size_t ws_size,
                              hipStream_t stream) {
    const float* Q = (const float*)d_in[0];
    const float* K = (const float*)d_in[1];
    const float* V = (const float*)d_in[2];
    float*       O = (float*)d_out;

    const int bh = in_sizes[0] / (SEQ * DH);   // batch * heads = 32
    dim3 grid(SEQ / MT, bh);
    dim3 block(NTHREADS);
    hipLaunchKernelGGL(flash_attn_wmma_f16, grid, block, 0, stream, Q, K, V, O);
}
